// LSHAttention_20255065768624
// MI455X (gfx1250) — matrix operand order repaired and checked by reading, not compiled, not run on hardware
//
#include <hip/hip_runtime.h>
#include <hip/hip_bf16.h>
#include <math.h>

// ---------------- problem constants (from reference) ----------------
#define B_   16
#define L_   8192
#define D_   64
#define R_   4
#define NB_  128   // n_buckets = L/BUCKET_LEN
#define BL_  64    // BUCKET_LEN
#define NH_  64    // n_buckets // 2 (rand projections per round)

typedef _Float16 v16h __attribute__((ext_vector_type(16)));
typedef float    v8f  __attribute__((ext_vector_type(8)));

// ======================================================================
// Kernel 0: inverse L2 norms of rand_matrix columns (over d, per b,r,n)
// rm layout: [B][D][R][NH]
// ======================================================================
__global__ __launch_bounds__(256) void rmnorm_kernel(
    const float* __restrict__ rm, float* __restrict__ inv_rm) {
  int idx = blockIdx.x * 256 + threadIdx.x;      // B*R*NH = 4096
  if (idx >= B_ * R_ * NH_) return;
  int n = idx & (NH_ - 1);
  int r = (idx / NH_) & (R_ - 1);
  int b = idx / (NH_ * R_);
  float ss = 0.f;
  for (int d = 0; d < D_; ++d) {
    float x = rm[(((size_t)b * D_ + d) * R_ + r) * NH_ + n];
    ss += x * x;
  }
  inv_rm[idx] = 1.0f / fmaxf(sqrtf(ss), 1e-12f);
}

// ======================================================================
// Kernel 1: LSH hash -> bucket id in [0,128) per (b,r,l)
// bucket layout: [B][R][L]
// grid: B * (L/256) blocks of 256 threads; one l per thread, rm chunk in LDS
// ======================================================================
__global__ __launch_bounds__(256) void hash_kernel(
    const float* __restrict__ query, const float* __restrict__ rm,
    const float* __restrict__ inv_rm, int* __restrict__ bucket) {
  __shared__ float chunk[D_ * NH_];              // 16 KB, [d][n] normalized
  int b = blockIdx.x >> 5;                       // 32 chunks per b
  int l = ((blockIdx.x & 31) << 8) + threadIdx.x;

  float q[D_];
  const float* src = query + ((size_t)b * L_ + l) * D_;
  #pragma unroll 8
  for (int d = 0; d < D_; ++d) q[d] = src[d];

  for (int r = 0; r < R_; ++r) {
    __syncthreads();
    for (int i = threadIdx.x; i < D_ * NH_; i += 256) {
      int d = i >> 6, n = i & 63;
      chunk[i] = rm[(((size_t)b * D_ + d) * R_ + r) * NH_ + n] *
                 inv_rm[((size_t)b * R_ + r) * NH_ + n];
    }
    __syncthreads();
    // argmax over concat([xR, -xR]); positive half wins ties (first index)
    float bestp = -3.4e38f, bestn = -3.4e38f;
    int ap = 0, an = 0;
    for (int n = 0; n < NH_; ++n) {
      float x = 0.f;
      #pragma unroll 8
      for (int d = 0; d < D_; ++d) x += q[d] * chunk[d * NH_ + n];
      if (x > bestp)  { bestp = x;  ap = n; }
      if (-x > bestn) { bestn = -x; an = n; }
    }
    int h = (bestn > bestp) ? (NH_ + an) : ap;
    bucket[((size_t)b * R_ + r) * L_ + l] = h;
  }
}

// ======================================================================
// Kernel 2: stable counting sort per (b,r): 128 threads, one per bucket id.
// Emits sorted_pos (positions in hash-sorted order) and sb (bucket per slot).
// Exactly reproduces argsort of unique keys h*L + pos.
// ======================================================================
__global__ __launch_bounds__(128) void sort_kernel(
    const int* __restrict__ bucket, int* __restrict__ sorted_pos,
    int* __restrict__ sb) {
  __shared__ unsigned char lb[L_];               // 8 KB
  __shared__ int counts[NB_];
  int br = blockIdx.x;                           // b*R + r
  int tid = threadIdx.x;                         // my bucket id
  size_t base = (size_t)br * L_;

  for (int i = tid; i < L_; i += 128)
    lb[i] = (unsigned char)bucket[base + i];
  __syncthreads();

  int cnt = 0;
  for (int i = 0; i < L_; ++i) cnt += (lb[i] == tid);
  counts[tid] = cnt;
  __syncthreads();
  if (tid == 0) {
    int run = 0;
    for (int i = 0; i < NB_; ++i) { int c = counts[i]; counts[i] = run; run += c; }
  }
  __syncthreads();

  int off = counts[tid];
  for (int i = 0; i < L_; ++i) {
    if (lb[i] == tid) {
      sorted_pos[base + off] = i;
      sb[base + off] = tid;
      ++off;
    }
  }
}

// ======================================================================
// Kernel 3: bucketed attention with WMMA.
// One block (256 thr = 8 waves) per (b, r, bucket-tile).
// S = (Q*1/8)(64x64) @ Knorm^T(64x128)  via v_wmma_f32_16x16x32_f16
// softmax+mask in f32 LDS, then P(64x128) @ Vlb(128x64) via WMMA.
// att written at ORIGINAL positions: att_ws[B][R][L][D]; lse_ws[B][R][L].
// ======================================================================
__global__ __launch_bounds__(256) void attn_kernel(
    const float* __restrict__ query, const float* __restrict__ value,
    const int* __restrict__ sorted_pos, const int* __restrict__ sb,
    float* __restrict__ att_ws, float* __restrict__ lse_ws) {
  // q_h(64x72) + k_h(128x72) f16, aliased by v_h(128x72) f16 after S is done
  __shared__ __align__(16) unsigned char smem_qkv[27648];
  __shared__ float s_f[64 * 132];                // S then P, f32 (33792 B)
  __shared__ int qpos[BL_], qbk[BL_];
  __shared__ int kpos[2 * BL_], kbk[2 * BL_];

  _Float16* q_h = (_Float16*)smem_qkv;           // 64 rows, stride 72
  _Float16* k_h = q_h + 64 * 72;                 // 128 rows, stride 72
  _Float16* v_h = (_Float16*)smem_qkv;           // 128 rows, stride 72 (alias)

  int blk = blockIdx.x;
  int b   = blk / (R_ * NB_);
  int rem = blk % (R_ * NB_);
  int r   = rem / NB_;
  int nbi = rem % NB_;
  int tid  = threadIdx.x;
  int lane = tid & 31;
  int wave = tid >> 5;
  int half = lane >> 4;                          // 0/1 lane-half
  int mrow = lane & 15;                          // A-matrix row within tile
  int ncol = lane & 15;                          // B/C/D column within tile

  size_t base = ((size_t)b * R_ + r) * L_;

  // ---- phase 1: gather slot indices (positions + bucket ids) ----
  if (tid < 64) {
    int slot = nbi * BL_ + tid;
    qpos[tid] = sorted_pos[base + slot];
    qbk[tid]  = sb[base + slot];
  } else if (tid < 192) {
    int j = tid - 64;                            // lookback key index 0..127
    int prev = (nbi + NB_ - 1) % NB_;
    int slot = (j < BL_) ? prev * BL_ + j : nbi * BL_ + (j - BL_);
    kpos[j] = sorted_pos[base + slot];
    kbk[j]  = sb[base + slot];
  }
  __syncthreads();

  // ---- phase 2: stage Q (pre-scaled by 1/sqrt(64)) and L2-normed K, f16 ----
  if (tid < 128) {
    int j = tid;
    const float* src = query + ((size_t)b * L_ + kpos[j]) * D_;
    float ss = 0.f;
    for (int d = 0; d < D_; ++d) { float x = src[d]; ss += x * x; }
    float inv = 1.0f / fmaxf(sqrtf(ss), 1e-12f);
    for (int d = 0; d < D_; ++d) k_h[j * 72 + d] = (_Float16)(src[d] * inv);
  } else if (tid < 192) {
    int i = tid - 128;
    const float* src = query + ((size_t)b * L_ + qpos[i]) * D_;
    for (int d = 0; d < D_; ++d) q_h[i * 72 + d] = (_Float16)(src[d] * 0.125f);
  }
  __syncthreads();

  // ---- phase 3: S = Qs @ K^T -> s_f (64x128). 32 tiles over 8 waves ----
  {
    int rt = wave & 3;                           // row-tile 0..3
    int ch = wave >> 2;                          // column half 0..1
    const _Float16* qrow = q_h + (16 * rt + mrow) * 72;
    v16h a0, a1;
    #pragma unroll
    for (int e = 0; e < 16; ++e) {
      // A fragment, contiguous k order: element e -> K=16*half+e (matches B)
      int k = 16 * half + e;
      a0[e] = qrow[k];
      a1[e] = qrow[32 + k];
    }
    #pragma unroll
    for (int c = 0; c < 4; ++c) {
      int ct = 4 * ch + c;
      const _Float16* krow = k_h + (16 * ct + ncol) * 72;  // B col = key row
      int kb = 16 * half;                        // B fragment, contiguous k order (matches A)
      v16h b0, b1;
      #pragma unroll
      for (int e = 0; e < 16; ++e) {
        b0[e] = krow[kb + e];
        b1[e] = krow[32 + kb + e];
      }
      v8f acc = {};
      acc = __builtin_amdgcn_wmma_f32_16x16x32_f16(false, a0, false, b0,
                                                   (short)0, acc, false, false);
      acc = __builtin_amdgcn_wmma_f32_16x16x32_f16(false, a1, false, b1,
                                                   (short)0, acc, false, false);
      #pragma unroll
      for (int v = 0; v < 8; ++v)                // C layout: M=v+8*half, N=ncol
        s_f[(16 * rt + v + 8 * half) * 132 + 16 * ct + ncol] = acc[v];
    }
  }
  __syncthreads();

  // ---- phase 4a: stage V lookback rows as f16 (overwrites q_h/k_h) ----
  {
    int j  = tid >> 1;
    int d0 = (tid & 1) * 32;
    const float* src = value + ((size_t)b * L_ + kpos[j]) * D_ + d0;
    for (int d = 0; d < 32; ++d) v_h[j * 72 + d0 + d] = (_Float16)src[d];
  }
  __syncthreads();

  // ---- phase 4b: masks + softmax per row; s_f overwritten with P (f32) ----
  if (tid < 64) {
    float* row = s_f + tid * 132;
    int qp = qpos[tid], qb = qbk[tid];
    float mx = -3.4e38f;
    for (int j = 0; j < 2 * BL_; ++j) {
      float v = row[j];
      if (qp == kpos[j]) v = -1e5f;              // self mask (applied last in ref)
      else if (qb != kbk[j]) v = -1e9f;          // cross-bucket mask
      row[j] = v;
      mx = fmaxf(mx, v);
    }
    float sum = 0.f;
    for (int j = 0; j < 2 * BL_; ++j) sum += expf(row[j] - mx);
    float lse = mx + logf(sum);
    for (int j = 0; j < 2 * BL_; ++j) row[j] = expf(row[j] - lse);
    lse_ws[base + qp] = lse;
  }
  __syncthreads();

  // ---- phase 5: att = P @ Vlb. 16 tiles over 8 waves (2 each) ----
  #pragma unroll
  for (int tt = 0; tt < 2; ++tt) {
    int t  = wave * 2 + tt;
    int rt = t >> 2, ct = t & 3;
    v8f acc = {};
    #pragma unroll
    for (int kc = 0; kc < 4; ++kc) {
      const float* prow = s_f + (16 * rt + mrow) * 132 + 32 * kc;
      v16h a, bb;
      #pragma unroll
      for (int e = 0; e < 16; ++e) {
        int k = 16 * half + e;
        a[e] = (_Float16)prow[k];                // P f32 -> f16 at fragment build
      }
      int kb = 16 * half;
      #pragma unroll
      for (int e = 0; e < 16; ++e)
        bb[e] = v_h[(32 * kc + kb + e) * 72 + 16 * ct + ncol];
      acc = __builtin_amdgcn_wmma_f32_16x16x32_f16(false, a, false, bb,
                                                   (short)0, acc, false, false);
    }
    #pragma unroll
    for (int v = 0; v < 8; ++v) {
      int i = 16 * rt + v + 8 * half;
      size_t p = (size_t)qpos[i];
      att_ws[(base + p) * D_ + 16 * ct + ncol] = acc[v];
    }
  }
}

// ======================================================================
// Kernel 4: per (b,r) max/sum of exp over L of lse (softmax over seq dim)
// ======================================================================
__global__ __launch_bounds__(256) void lsered_kernel(
    const float* __restrict__ lse, float* __restrict__ mws,
    float* __restrict__ zws) {
  __shared__ float red[256];
  int br = blockIdx.x;
  int tid = threadIdx.x;
  const float* p = lse + (size_t)br * L_;
  float mx = -3.4e38f;
  for (int i = tid; i < L_; i += 256) mx = fmaxf(mx, p[i]);
  red[tid] = mx; __syncthreads();
  for (int s = 128; s > 0; s >>= 1) {
    if (tid < s) red[tid] = fmaxf(red[tid], red[tid + s]);
    __syncthreads();
  }
  float m = red[0]; __syncthreads();
  float sm = 0.f;
  for (int i = tid; i < L_; i += 256) sm += expf(p[i] - m);
  red[tid] = sm; __syncthreads();
  for (int s = 128; s > 0; s >>= 1) {
    if (tid < s) red[tid] += red[tid + s];
    __syncthreads();
  }
  if (tid == 0) { mws[br] = m; zws[br] = red[0]; }
}

// ======================================================================
// Kernel 5: out[b,l,d] = sum_r att[b,r,l,d] * exp(lse[b,r,l]-m)/Z
// ======================================================================
__global__ __launch_bounds__(256) void combine_kernel(
    const float* __restrict__ att, const float* __restrict__ lse,
    const float* __restrict__ mws, const float* __restrict__ zws,
    float* __restrict__ out) {
  size_t o = (size_t)blockIdx.x * 256 + threadIdx.x;
  if (o >= (size_t)B_ * L_ * D_) return;
  size_t ld = o / D_;
  int d = (int)(o & (D_ - 1));
  int b = (int)(ld / L_);
  int l = (int)(ld % L_);
  float acc = 0.f;
  #pragma unroll
  for (int r = 0; r < R_; ++r) {
    size_t br = (size_t)b * R_ + r;
    float w = expf(lse[br * L_ + l] - mws[br]) / zws[br];
    acc += att[(br * L_ + l) * D_ + d] * w;
  }
  out[o] = acc;
}

// ======================================================================
// launch
// ======================================================================
extern "C" void kernel_launch(void* const* d_in, const int* in_sizes, int n_in,
                              void* d_out, int out_size, void* d_ws,
                              size_t ws_size, hipStream_t stream) {
  const float* query = (const float*)d_in[0];
  const float* value = (const float*)d_in[1];
  const float* rm    = (const float*)d_in[2];
  float* out = (float*)d_out;

  char* ws = (char*)d_ws;
  size_t off = 0;
  auto carve = [&](size_t bytes) -> char* {
    char* p = ws + off;
    off = (off + bytes + 255) & ~(size_t)255;
    return p;
  };
  float* inv_rm     = (float*)carve((size_t)B_ * R_ * NH_ * 4);
  int*   bucket     = (int*)  carve((size_t)B_ * R_ * L_ * 4);
  int*   sorted_pos = (int*)  carve((size_t)B_ * R_ * L_ * 4);
  int*   sb         = (int*)  carve((size_t)B_ * R_ * L_ * 4);
  float* lse_ws     = (float*)carve((size_t)B_ * R_ * L_ * 4);
  float* mws        = (float*)carve((size_t)B_ * R_ * 4);
  float* zws        = (float*)carve((size_t)B_ * R_ * 4);
  float* att_ws     = (float*)carve((size_t)B_ * R_ * L_ * D_ * 4);
  (void)ws_size; (void)in_sizes; (void)n_in; (void)out_size;

  rmnorm_kernel<<<dim3((B_ * R_ * NH_ + 255) / 256), dim3(256), 0, stream>>>(
      rm, inv_rm);
  hash_kernel<<<dim3(B_ * (L_ / 256)), dim3(256), 0, stream>>>(
      query, rm, inv_rm, bucket);
  sort_kernel<<<dim3(B_ * R_), dim3(128), 0, stream>>>(
      bucket, sorted_pos, sb);
  attn_kernel<<<dim3(B_ * R_ * NB_), dim3(256), 0, stream>>>(
      query, value, sorted_pos, sb, att_ws, lse_ws);
  lsered_kernel<<<dim3(B_ * R_), dim3(256), 0, stream>>>(lse_ws, mws, zws);
  combine_kernel<<<dim3((unsigned)(((size_t)B_ * L_ * D_ + 255) / 256)),
                   dim3(256), 0, stream>>>(att_ws, lse_ws, mws, zws, out);
}
